// HPTModel_42382737276994
// MI455X (gfx1250) — matrix operand order repaired and checked by reading, not compiled, not run on hardware
//
#include <hip/hip_runtime.h>
#include <hip/hip_bf16.h>
#include <math.h>

#define BB 2
#define SS 4096
#define DD 768
#define HH 12
#define LLAY 2
#define DHD 64
#define CCH 256
#define GGL 64
#define FFD 3072
#define MSEL 200
#define MPAD 208
#define XX 4
#define NCC 16
#define KEY 832          // 3*C + G
#define BSTOT (BB*SS)
#define SCALEF 0.125f    // 1/sqrt(64)

typedef _Float16 h16;
typedef __attribute__((ext_vector_type(16))) _Float16 v16h;
typedef __attribute__((ext_vector_type(8)))  _Float16 v8h;
typedef __attribute__((ext_vector_type(8)))  float    v8f;

// ---------------- WMMA helpers ----------------
__device__ __forceinline__ v8f wmma_f16(v16h a, v16h b, v8f c) {
  return __builtin_amdgcn_wmma_f32_16x16x32_f16(false, a, false, b, (short)0, c, false, false);
}

// A fragment (16x32 f16): lane holds row lane%16; elem e <-> K = (lane/16)*16 + e (contiguous)
__device__ __forceinline__ v16h load_a_frag(const h16* p, int lane) {
  const h16* q = p + ((lane >> 4) << 4);
  v8h lo = *(const v8h*)q;
  v8h hi = *(const v8h*)(q + 8);
  v16h r;
#pragma unroll
  for (int i = 0; i < 8; ++i) { r[i] = lo[i]; r[i + 8] = hi[i]; }
  return r;
}

// B fragment (32x16 f16): lane holds col lane%16; elem e <-> K = (lane/16)*16 + e (contiguous)
__device__ __forceinline__ v16h load_b_frag(const h16* p, int lane) {
  const h16* q = p + ((lane >> 4) << 4);
  v8h lo = *(const v8h*)q;
  v8h hi = *(const v8h*)(q + 8);
  v16h r;
#pragma unroll
  for (int i = 0; i < 8; ++i) { r[i] = lo[i]; r[i + 8] = hi[i]; }
  return r;
}

// B fragment, caller already folded the (lane/16)*16 K-offset into the pointer
__device__ __forceinline__ v16h load_b16d(const h16* p) {
  v8h lo = *(const v8h*)p;
  v8h hi = *(const v8h*)(p + 8);
  v16h r;
#pragma unroll
  for (int i = 0; i < 8; ++i) { r[i] = lo[i]; r[i + 8] = hi[i]; }
  return r;
}

__device__ __forceinline__ float warp_sum(float v) {
#pragma unroll
  for (int o = 16; o; o >>= 1) v += __shfl_xor(v, o, 32);
  return v;
}
__device__ __forceinline__ float warp_max(float v) {
#pragma unroll
  for (int o = 16; o; o >>= 1) v = fmaxf(v, __shfl_xor(v, o, 32));
  return v;
}
__device__ __forceinline__ float gelu_f(float x) {
  float x3 = x * x * x;
  return 0.5f * x * (1.f + tanhf(0.7978845608028654f * (x + 0.044715f * x3)));
}
__device__ __forceinline__ float bce_f(float p, float t) {
  return -(t * logf(p + 1e-12f) + (1.f - t) * logf(1.f - p + 1e-12f));
}

// ---------------- conversion / transpose ----------------
// W [K,N] f32 -> Wt [N,K] f16
__global__ __launch_bounds__(256)
void k_w2h_t(const float* __restrict__ W, h16* __restrict__ Wt, int K, int N) {
  long i = (long)blockIdx.x * blockDim.x + threadIdx.x;
  if (i >= (long)K * N) return;
  int n = (int)(i / K), k = (int)(i % K);
  Wt[i] = (h16)W[(long)k * N + n];
}

// V [B*S, D(h,dh)] f16 -> Vt [B,H,DH,S] f16
__global__ __launch_bounds__(256)
void k_vtrans(const h16* __restrict__ vh, h16* __restrict__ vt) {
  long i = (long)blockIdx.x * blockDim.x + threadIdx.x;
  if (i >= (long)BB * HH * DHD * SS) return;
  int s = (int)(i % SS);
  long r = i / SS;
  int dh = (int)(r % DHD); r /= DHD;
  int h = (int)(r % HH);
  int b = (int)(r / HH);
  vt[i] = vh[((long)(b * SS + s)) * DD + h * DHD + dh];
}

// ---------------- embedding + LN / residual + LN ----------------
__global__ __launch_bounds__(256)
void k_embed_ln(const int* __restrict__ ids, const float* __restrict__ emb,
                const float* __restrict__ pos, const float* __restrict__ g,
                const float* __restrict__ be, float* __restrict__ x, h16* __restrict__ xh) {
  __shared__ float red[256];
  int bsn = blockIdx.x;
  int s = bsn % SS;
  int t = threadIdx.x;
  const float* e = emb + (long)ids[bsn] * DD;
  const float* p = pos + (long)s * DD;
  float v[3]; float loc = 0.f;
#pragma unroll
  for (int j = 0; j < 3; ++j) { int d = t * 3 + j; v[j] = e[d] + p[d]; loc += v[j]; }
  red[t] = loc; __syncthreads();
  for (int o = 128; o; o >>= 1) { if (t < o) red[t] += red[t + o]; __syncthreads(); }
  float mean = red[0] * (1.f / DD); __syncthreads();
  loc = 0.f;
#pragma unroll
  for (int j = 0; j < 3; ++j) { float d2 = v[j] - mean; loc += d2 * d2; }
  red[t] = loc; __syncthreads();
  for (int o = 128; o; o >>= 1) { if (t < o) red[t] += red[t + o]; __syncthreads(); }
  float rstd = rsqrtf(red[0] * (1.f / DD) + 1e-5f);
#pragma unroll
  for (int j = 0; j < 3; ++j) {
    int d = t * 3 + j;
    float o2 = (v[j] - mean) * rstd * g[d] + be[d];
    long off = (long)bsn * DD + d;
    x[off] = o2; xh[off] = (h16)o2;
  }
}

__global__ __launch_bounds__(256)
void k_addln(float* __restrict__ x, const float* __restrict__ y,
             const float* __restrict__ g, const float* __restrict__ be,
             h16* __restrict__ xh) {
  __shared__ float red[256];
  int bsn = blockIdx.x;
  int t = threadIdx.x;
  long base = (long)bsn * DD;
  float v[3]; float loc = 0.f;
#pragma unroll
  for (int j = 0; j < 3; ++j) { int d = t * 3 + j; v[j] = x[base + d] + y[base + d]; loc += v[j]; }
  red[t] = loc; __syncthreads();
  for (int o = 128; o; o >>= 1) { if (t < o) red[t] += red[t + o]; __syncthreads(); }
  float mean = red[0] * (1.f / DD); __syncthreads();
  loc = 0.f;
#pragma unroll
  for (int j = 0; j < 3; ++j) { float d2 = v[j] - mean; loc += d2 * d2; }
  red[t] = loc; __syncthreads();
  for (int o = 128; o; o >>= 1) { if (t < o) red[t] += red[t + o]; __syncthreads(); }
  float rstd = rsqrtf(red[0] * (1.f / DD) + 1e-5f);
#pragma unroll
  for (int j = 0; j < 3; ++j) {
    int d = t * 3 + j;
    float o2 = (v[j] - mean) * rstd * g[d] + be[d];
    x[base + d] = o2; xh[base + d] = (h16)o2;
  }
}

// ---------------- blocked WMMA GEMM: wave computes 16x64, A-frag reused 4x ----------------
__device__ __forceinline__ void store_tile(v8f c, int row0, int col, int lh,
                                           const float* __restrict__ bias,
                                           float* __restrict__ Cf, h16* __restrict__ Ch,
                                           int ldc, int M, int N, int act) {
  float bv = bias ? bias[col < N ? col : N - 1] : 0.f;
#pragma unroll
  for (int r = 0; r < 8; ++r) {
    int rw = row0 + (lh << 3) + r;
    if (rw < M && col < N) {
      float v = c[r] + bv;
      if (act == 1) v = gelu_f(v);
      else if (act == 2) v = fmaxf(v, 0.f);
      long o = (long)rw * ldc + col;
      if (Cf) Cf[o] = v;
      if (Ch) Ch[o] = (h16)v;
    }
  }
}

__global__ __launch_bounds__(32)
void k_gemm(const h16* __restrict__ A, int lda, const h16* __restrict__ Bt, int ldb,
            const float* __restrict__ bias, float* __restrict__ Cf, h16* __restrict__ Ch,
            int ldc, int M, int N, int K, int act, int nblk) {
  int bx = blockIdx.x % nblk;           // 64-wide N block
  int by = blockIdx.x / nblk;
  int lane = threadIdx.x;
  int l16 = lane & 15, lh = lane >> 4;
  int row0 = by * 16, col0 = bx * 64;
  int arow = row0 + l16; if (arow >= M) arow = M - 1;
  const h16* ap = A + (long)arow * lda;
  int bc0 = col0 + l16;      if (bc0 >= N) bc0 = N - 1;
  int bc1 = col0 + 16 + l16; if (bc1 >= N) bc1 = N - 1;
  int bc2 = col0 + 32 + l16; if (bc2 >= N) bc2 = N - 1;
  int bc3 = col0 + 48 + l16; if (bc3 >= N) bc3 = N - 1;
  const h16* bp0 = Bt + (long)bc0 * ldb;
  const h16* bp1 = Bt + (long)bc1 * ldb;
  const h16* bp2 = Bt + (long)bc2 * ldb;
  const h16* bp3 = Bt + (long)bc3 * ldb;
  v8f c0 = {}, c1 = {}, c2 = {}, c3 = {};
  for (int k0 = 0; k0 < K; k0 += 32) {
    __builtin_prefetch(ap + k0 + 128, 0, 1);
    v16h a = load_a_frag(ap + k0, lane);
    c0 = wmma_f16(a, load_b_frag(bp0 + k0, lane), c0);
    c1 = wmma_f16(a, load_b_frag(bp1 + k0, lane), c1);
    c2 = wmma_f16(a, load_b_frag(bp2 + k0, lane), c2);
    c3 = wmma_f16(a, load_b_frag(bp3 + k0, lane), c3);
  }
  store_tile(c0, row0, col0 + l16,      lh, bias, Cf, Ch, ldc, M, N, act);
  store_tile(c1, row0, col0 + 16 + l16, lh, bias, Cf, Ch, ldc, M, N, act);
  store_tile(c2, row0, col0 + 32 + l16, lh, bias, Cf, Ch, ldc, M, N, act);
  store_tile(c3, row0, col0 + 48 + l16, lh, bias, Cf, Ch, ldc, M, N, act);
}

// ---------------- fused local attention: scores -> softmax (LDS) -> P*V ----------------
__global__ __launch_bounds__(32)
void k_local_attn(const h16* __restrict__ qh, const h16* __restrict__ kh,
                  const h16* __restrict__ vt, const float* __restrict__ am,
                  const float* __restrict__ gm, h16* __restrict__ attn_h) {
  __shared__ __align__(16) float sc[16 * KEY];   // 53 KB score tile
  int id = blockIdx.x;
  int mt = id & 15;
  int nc = (id >> 4) & 15;
  int h  = (id >> 8) % HH;
  int b  = id / (16 * 16 * HH);
  int lane = threadIdx.x, l16 = lane & 15, lh = lane >> 4;
  int qrow = nc * CCH + mt * 16 + l16;
  const h16* qp = qh + ((long)(b * SS + qrow)) * DD + h * DHD;
  v16h qa0 = load_a_frag(qp, lane);        // loop-invariant Q fragments
  v16h qa1 = load_a_frag(qp + 32, lane);

  // phase 1: scores (WMMA) -> LDS
  for (int nt = 0; nt < KEY / 16; ++nt) {
    int key = nt * 16 + l16;
    const h16* kp;
    float mvalid;
    if (nt < 48) {
      int sk = (nc - 1) * CCH + key;
      bool inb = (sk >= 0) && (sk < SS);
      int skc = inb ? sk : 0;
      kp = kh + ((long)(b * SS + skc)) * DD + h * DHD;
      mvalid = inb ? am[b * SS + sk] : 0.f;
    } else {
      int gk = key - 3 * CCH;
      kp = kh + ((long)(b * SS + gk)) * DD + h * DHD;
      mvalid = am[b * SS + gk] * gm[b * SS + gk];
    }
    v8f c = {};
    c = wmma_f16(qa0, load_b_frag(kp, lane), c);
    c = wmma_f16(qa1, load_b_frag(kp + 32, lane), c);
#pragma unroll
    for (int r = 0; r < 8; ++r) {
      float v = (mvalid > 0.f) ? c[r] * SCALEF : -1e9f;
      sc[((lh << 3) + r) * KEY + nt * 16 + l16] = v;
    }
  }
  __syncthreads();

  // phase 2: row softmax, overwrite in place as f16 P (read-regs-then-write, per row)
  h16* ph = (h16*)sc;
  for (int r = 0; r < 16; ++r) {
    float vv[KEY / 32];
    float m = -1e30f;
#pragma unroll
    for (int i = 0; i < KEY / 32; ++i) { vv[i] = sc[r * KEY + lane + i * 32]; m = fmaxf(m, vv[i]); }
    m = warp_max(m);
    float sum = 0.f;
#pragma unroll
    for (int i = 0; i < KEY / 32; ++i) { vv[i] = __expf(vv[i] - m); sum += vv[i]; }
    sum = warp_sum(sum);
    float inv = 1.f / sum;
    __builtin_amdgcn_wave_barrier();
#pragma unroll
    for (int i = 0; i < KEY / 32; ++i) ph[r * KEY + lane + i * 32] = (h16)(vv[i] * inv);
    __builtin_amdgcn_wave_barrier();
  }
  __syncthreads();

  // phase 3: out = P @ V  (A frag from LDS loaded once, reused across 4 DH tiles)
  const h16* prow = ph + (long)l16 * KEY;
  long vbase = (((long)(b * HH + h)) * DHD + l16) * SS;
  const h16* vr0 = vt + vbase;              // dh = l16
  const h16* vr1 = vt + vbase + 16L * SS;   // dh = 16 + l16
  const h16* vr2 = vt + vbase + 32L * SS;
  const h16* vr3 = vt + vbase + 48L * SS;
  v8f c0 = {}, c1 = {}, c2 = {}, c3 = {};
  for (int ks = 0; ks < KEY / 32; ++ks) {
    v16h a = load_a_frag(prow + ks * 32, lane);
    int kd = ks * 32 + (lh << 4);
    long off;
    if (ks < 24) {
      int s0 = (nc - 1) * CCH + kd;
      if (s0 < 0) s0 = 0;
      if (s0 > SS - 16) s0 = SS - 16;
      off = s0;
    } else {
      off = kd - 3 * CCH;
    }
    c0 = wmma_f16(a, load_b16d(vr0 + off), c0);
    c1 = wmma_f16(a, load_b16d(vr1 + off), c1);
    c2 = wmma_f16(a, load_b16d(vr2 + off), c2);
    c3 = wmma_f16(a, load_b16d(vr3 + off), c3);
  }
#pragma unroll
  for (int r = 0; r < 8; ++r) {
    int qs = nc * CCH + mt * 16 + (lh << 3) + r;
    long ob = ((long)(b * SS + qs)) * DD + h * DHD;
    attn_h[ob + l16]      = (h16)c0[r];
    attn_h[ob + 16 + l16] = (h16)c1[r];
    attn_h[ob + 32 + l16] = (h16)c2[r];
    attn_h[ob + 48 + l16] = (h16)c3[r];
  }
}

// ---------------- global-token attention (overwrites first G rows) ----------------
__global__ __launch_bounds__(32)
void k_gscore(const h16* __restrict__ qh, const h16* __restrict__ kh,
              const float* __restrict__ am, float* __restrict__ Sg) {
  int id = blockIdx.x;
  int nt = id % (SS / 64);
  int mt = (id / (SS / 64)) % 4;
  int h  = (id / (SS / 64 * 4)) % HH;
  int b  = id / (SS / 64 * 4 * HH);
  int lane = threadIdx.x, l16 = lane & 15, lh = lane >> 4;
  const h16* ap = qh + ((long)(b * SS + mt * 16 + l16)) * DD + h * DHD;
  v16h qa0 = load_a_frag(ap, lane);
  v16h qa1 = load_a_frag(ap + 32, lane);
  int k0 = nt * 64 + l16;
  const h16* bp0 = kh + ((long)(b * SS + k0)) * DD + h * DHD;
  const h16* bp1 = kh + ((long)(b * SS + k0 + 16)) * DD + h * DHD;
  const h16* bp2 = kh + ((long)(b * SS + k0 + 32)) * DD + h * DHD;
  const h16* bp3 = kh + ((long)(b * SS + k0 + 48)) * DD + h * DHD;
  v8f c0 = {}, c1 = {}, c2 = {}, c3 = {};
  c0 = wmma_f16(qa0, load_b_frag(bp0, lane), c0);
  c1 = wmma_f16(qa0, load_b_frag(bp1, lane), c1);
  c2 = wmma_f16(qa0, load_b_frag(bp2, lane), c2);
  c3 = wmma_f16(qa0, load_b_frag(bp3, lane), c3);
  c0 = wmma_f16(qa1, load_b_frag(bp0 + 32, lane), c0);
  c1 = wmma_f16(qa1, load_b_frag(bp1 + 32, lane), c1);
  c2 = wmma_f16(qa1, load_b_frag(bp2 + 32, lane), c2);
  c3 = wmma_f16(qa1, load_b_frag(bp3 + 32, lane), c3);
  long rowb = ((long)(b * HH + h)) * GGL;
#pragma unroll
  for (int j = 0; j < 4; ++j) {
    int key = nt * 64 + j * 16 + l16;
    float mv = am[b * SS + key];
    v8f c = (j == 0) ? c0 : (j == 1) ? c1 : (j == 2) ? c2 : c3;
#pragma unroll
    for (int r = 0; r < 8; ++r) {
      int row = mt * 16 + (lh << 3) + r;
      Sg[(rowb + row) * SS + key] = (mv > 0.f) ? c[r] * SCALEF : -1e9f;
    }
  }
}

__global__ __launch_bounds__(32)
void k_gsoftmax(const float* __restrict__ Sg, h16* __restrict__ Pg) {
  int row = blockIdx.x;
  int lane = threadIdx.x;
  const float* sp = Sg + (long)row * SS;
  float m = -1e30f;
  for (int i = lane; i < SS; i += 32) m = fmaxf(m, sp[i]);
  m = warp_max(m);
  float sum = 0.f;
  for (int i = lane; i < SS; i += 32) sum += __expf(sp[i] - m);
  sum = warp_sum(sum);
  float inv = 1.f / sum;
  h16* pp = Pg + (long)row * SS;
  for (int i = lane; i < SS; i += 32) pp[i] = (h16)(__expf(sp[i] - m) * inv);
}

__global__ __launch_bounds__(32)
void k_gout(const h16* __restrict__ Pg, const h16* __restrict__ vt, h16* __restrict__ attn_h) {
  int id = blockIdx.x;                 // B*H*4
  int mt = id % 4;
  int h  = (id / 4) % HH;
  int b  = id / (4 * HH);
  int lane = threadIdx.x, l16 = lane & 15, lh = lane >> 4;
  const h16* ap = Pg + (((long)(b * HH + h)) * GGL + mt * 16 + l16) * SS;
  long vbase = (((long)(b * HH + h)) * DHD + l16) * SS;
  const h16* vr0 = vt + vbase;
  const h16* vr1 = vt + vbase + 16L * SS;
  const h16* vr2 = vt + vbase + 32L * SS;
  const h16* vr3 = vt + vbase + 48L * SS;
  v8f c0 = {}, c1 = {}, c2 = {}, c3 = {};
  for (int ks = 0; ks < SS / 32; ++ks) {
    v16h a = load_a_frag(ap + ks * 32, lane);
    c0 = wmma_f16(a, load_b_frag(vr0 + ks * 32, lane), c0);
    c1 = wmma_f16(a, load_b_frag(vr1 + ks * 32, lane), c1);
    c2 = wmma_f16(a, load_b_frag(vr2 + ks * 32, lane), c2);
    c3 = wmma_f16(a, load_b_frag(vr3 + ks * 32, lane), c3);
  }
#pragma unroll
  for (int r = 0; r < 8; ++r) {
    int qs = mt * 16 + (lh << 3) + r;
    long ob = ((long)(b * SS + qs)) * DD + h * DHD;
    attn_h[ob + l16]      = (h16)c0[r];
    attn_h[ob + 16 + l16] = (h16)c1[r];
    attn_h[ob + 32 + l16] = (h16)c2[r];
    attn_h[ob + 48 + l16] = (h16)c3[r];
  }
}

// ---------------- heads / losses ----------------
__global__ __launch_bounds__(256)
void k_heads(const float* __restrict__ x, const float* __restrict__ Wa,
             const float* __restrict__ ba, const float* __restrict__ Wsp,
             const float* __restrict__ bsp, float* __restrict__ pe, float* __restrict__ ps) {
  int idx = blockIdx.x * 8 + (threadIdx.x >> 5);
  if (idx >= BB * SS * 5) return;
  int j = idx % 5;
  int bsn = idx / 5;
  int lane = threadIdx.x & 31;
  const float* hr = x + (long)bsn * DD;
  float acc = 0.f;
  if (j < 3) { for (int d = lane; d < DD; d += 32) acc += hr[d] * Wa[d * 3 + j]; }
  else { int jj = j - 3; for (int d = lane; d < DD; d += 32) acc += hr[d] * Wsp[d * 2 + jj]; }
  acc = warp_sum(acc);
  if (lane == 0) {
    if (j < 3) pe[(long)bsn * 3 + j] = 1.f / (1.f + __expf(-(acc + ba[j])));
    else       ps[(long)bsn * 2 + (j - 3)] = 1.f / (1.f + __expf(-(acc + bsp[j - 3])));
  }
}

__global__ __launch_bounds__(256)
void k_count_es(const float* __restrict__ mlh, const float* __restrict__ mas, float* acc) {
  int i0 = blockIdx.x * blockDim.x + threadIdx.x;
  int stride = gridDim.x * blockDim.x;
  float ne = 0, nze = 0, ns = 0, nzs = 0;
  for (int k = i0; k < BB * SS * 3; k += stride) { float m = mlh[k]; if (m != 0.f) { ne += 1.f; if (m > 0.f) nze += 1.f; } }
  for (int k = i0; k < BB * SS * 2; k += stride) { float m = mas[k]; if (m != 0.f) { ns += 1.f; if (m > 0.f) nzs += 1.f; } }
  if (ne) atomicAdd(&acc[0], ne);
  if (nze) atomicAdd(&acc[1], nze);
  if (ns) atomicAdd(&acc[2], ns);
  if (nzs) atomicAdd(&acc[3], nzs);
}

__global__ __launch_bounds__(256)
void k_loss_es(const float* __restrict__ pe, const float* __restrict__ ps,
               const float* __restrict__ mlh, const float* __restrict__ mas, float* acc) {
  float n_e = acc[0], nz_e = acc[1], n_s = acc[2], nz_s = acc[3];
  float xwe = n_e / (nz_e * 2.f), ywe = n_e / ((n_e - nz_e) * 2.f);
  float xws = n_s / (nz_s * 2.f), yws = n_s / ((n_s - nz_s) * 2.f);
  int i0 = blockIdx.x * blockDim.x + threadIdx.x;
  int stride = gridDim.x * blockDim.x;
  float se = 0, ss2 = 0;
  for (int k = i0; k < BB * SS * 3; k += stride) {
    float m = mlh[k];
    if (m != 0.f) { float t = (m + 1.f) * 0.5f; float w = (t == 1.f) ? xwe : ywe; se += w * bce_f(pe[k], t); }
  }
  for (int k = i0; k < BB * SS * 2; k += stride) {
    float m = mas[k];
    if (m != 0.f) { float t = (m + 1.f) * 0.5f; float w = (t == 1.f) ? xws : yws; ss2 += w * bce_f(ps[k], t); }
  }
  if (se) atomicAdd(&acc[4], se);
  if (ss2) atomicAdd(&acc[5], ss2);
}

__global__ void k_order(const float* __restrict__ mh, int* __restrict__ order, float* __restrict__ mask_r) {
  int b = blockIdx.x;
  if (threadIdx.x != 0) return;
  int cnt = 0;
  for (int s = 0; s < SS && cnt < MSEL; ++s) if (mh[b * SS + s] > 0.f) order[b * MPAD + cnt++] = s;
  for (int s = 0; s < SS && cnt < MSEL; ++s) if (!(mh[b * SS + s] > 0.f)) order[b * MPAD + cnt++] = s;
  for (int m = MSEL; m < MPAD; ++m) order[b * MPAD + m] = 0;
  for (int m = 0; m < MPAD; ++m) mask_r[b * MPAD + m] = (m < MSEL) ? mh[b * SS + order[b * MPAD + m]] : 0.f;
}

__global__ __launch_bounds__(256)
void k_gather(const float* __restrict__ x, const int* __restrict__ order, h16* __restrict__ hrh) {
  int bm = blockIdx.x;
  int b = bm / MPAD, m = bm % MPAD;
  int t = threadIdx.x;
  int o = order[bm];
#pragma unroll
  for (int j = 0; j < 3; ++j) {
    int d = t * 3 + j;
    h16 v = (m < MSEL) ? (h16)x[((long)(b * SS + o)) * DD + d] : (h16)0.f;
    hrh[(long)bm * DD + d] = v;
  }
}

// pred_cond logits: pc[b,c,q] = ht[b,c,:] . hr[b,q,:]
__global__ __launch_bounds__(32)
void k_pc(const h16* __restrict__ ht, const h16* __restrict__ hrh, float* __restrict__ pc) {
  int id = blockIdx.x;
  int qt = id % 13;
  int ct = (id / 13) % 13;
  int b = id / 169;
  int lane = threadIdx.x, l16 = lane & 15, lh = lane >> 4;
  const h16* ap = ht + ((long)(b * MPAD + ct * 16 + l16)) * DD;
  const h16* bp = hrh + ((long)(b * MPAD + qt * 16 + l16)) * DD;
  v8f c = {};
  for (int k0 = 0; k0 < DD; k0 += 32) {
    v16h a  = load_a_frag(ap + k0, lane);
    v16h bf = load_b_frag(bp + k0, lane);
    c = wmma_f16(a, bf, c);
  }
#pragma unroll
  for (int r = 0; r < 8; ++r) {
    int cr = ct * 16 + (lh << 3) + r;
    int qc = qt * 16 + l16;
    pc[((long)b * MPAD + cr) * MPAD + qc] = c[r];
  }
}

__global__ __launch_bounds__(256)
void k_count_cond(const float* __restrict__ mask_r, const int* __restrict__ order,
                  const float* __restrict__ mlc, float* acc) {
  int i0 = blockIdx.x * blockDim.x + threadIdx.x;
  int stride = gridDim.x * blockDim.x;
  float n = 0, nz = 0;
  for (int k = i0; k < BB * MSEL * MSEL; k += stride) {
    int q = k % MSEL;
    int c = (k / MSEL) % MSEL;
    int b = k / (MSEL * MSEL);
    float sel = mask_r[b * MPAD + c] * mask_r[b * MPAD + q];
    if (sel != 0.f) {
      int sq = order[b * MPAD + q], sc2 = order[b * MPAD + c];
      float t = 0.f;
      for (int xx = 0; xx < XX; ++xx)
        t += mlc[(((long)b * SS + sq) * XX + xx) * 2 + 1] * mlc[(((long)b * SS + sc2) * XX + xx) * 2 + 0];
      n += 1.f;
      if (t != 0.f) nz += 1.f;
    }
  }
  if (n) atomicAdd(&acc[6], n);
  if (nz) atomicAdd(&acc[7], nz);
}

__global__ __launch_bounds__(256)
void k_loss_cond(const float* __restrict__ pc, const float* __restrict__ mask_r,
                 const int* __restrict__ order, const float* __restrict__ mlc, float* acc) {
  float n = acc[6], nz = acc[7];
  float xw = n / (nz * 2.f), yw = n / ((n - nz) * 2.f);
  int i0 = blockIdx.x * blockDim.x + threadIdx.x;
  int stride = gridDim.x * blockDim.x;
  float s = 0;
  for (int k = i0; k < BB * MSEL * MSEL; k += stride) {
    int q = k % MSEL;
    int c = (k / MSEL) % MSEL;
    int b = k / (MSEL * MSEL);
    float sel = mask_r[b * MPAD + c] * mask_r[b * MPAD + q];
    if (sel != 0.f) {
      int sq = order[b * MPAD + q], sc2 = order[b * MPAD + c];
      float t = 0.f;
      for (int xx = 0; xx < XX; ++xx)
        t += mlc[(((long)b * SS + sq) * XX + xx) * 2 + 1] * mlc[(((long)b * SS + sc2) * XX + xx) * 2 + 0];
      float p = 1.f / (1.f + __expf(-pc[((long)b * MPAD + c) * MPAD + q]));
      float w = (t == 1.f) ? xw : yw;
      s += w * bce_f(p, t);
    }
  }
  if (s) atomicAdd(&acc[8], s);
}

__global__ void k_zero(float* acc) {
  if (blockIdx.x == 0 && threadIdx.x < 16) acc[threadIdx.x] = 0.f;
}

__global__ void k_finalize(const float* acc, float* out) {
  if (blockIdx.x || threadIdx.x) return;
  float le = acc[4] / acc[0];
  float ls = acc[5] / acc[2];
  float lc = acc[8] / acc[6];
  out[0] = le + ls + 0.f * lc;
  out[1] = le;
  out[2] = ls;
  out[3] = lc;
}

// ---------------- host orchestration ----------------
struct ParamPtrs {
  const float *emb, *pos, *ln_g, *ln_b;
  const float *Wq[2], *bq[2], *Wk[2], *bk[2], *Wv[2], *bv[2], *Wo[2], *bo[2];
  const float *ln1g[2], *ln1b[2], *W1[2], *b1[2], *W2[2], *b2[2], *ln2g[2], *ln2b[2];
  const float *Wa, *ba, *Wsp, *bsp, *Wc, *bc;
  const int* ids;
  const float *am, *gm, *mh, *mlh, *mas, *mlc;
};

static void read_layer_sorted(void* const* d, int& i, ParamPtrs& p, int l) {
  p.W1[l] = (const float*)d[i++]; p.W2[l] = (const float*)d[i++];
  p.Wk[l] = (const float*)d[i++]; p.Wo[l] = (const float*)d[i++];
  p.Wq[l] = (const float*)d[i++]; p.Wv[l] = (const float*)d[i++];
  p.b1[l] = (const float*)d[i++]; p.b2[l] = (const float*)d[i++];
  p.bk[l] = (const float*)d[i++]; p.bo[l] = (const float*)d[i++];
  p.bq[l] = (const float*)d[i++]; p.bv[l] = (const float*)d[i++];
  p.ln1b[l] = (const float*)d[i++]; p.ln1g[l] = (const float*)d[i++];
  p.ln2b[l] = (const float*)d[i++]; p.ln2g[l] = (const float*)d[i++];
}
static void read_params_sorted(void* const* d, int& i, ParamPtrs& p) {
  p.Wa = (const float*)d[i++]; p.Wc = (const float*)d[i++]; p.Wsp = (const float*)d[i++];
  p.ba = (const float*)d[i++]; p.bc = (const float*)d[i++]; p.bsp = (const float*)d[i++];
  p.emb = (const float*)d[i++];
  for (int l = 0; l < LLAY; ++l) read_layer_sorted(d, i, p, l);
  p.ln_b = (const float*)d[i++]; p.ln_g = (const float*)d[i++]; p.pos = (const float*)d[i++];
}

extern "C" void kernel_launch(void* const* d_in, const int* in_sizes, int n_in,
                              void* d_out, int out_size, void* d_ws, size_t ws_size,
                              hipStream_t stream) {
  (void)out_size; (void)ws_size; (void)n_in;
  const long EMBSZ = (long)50282 * DD;

  ParamPtrs p;
  if (in_sizes[0] == (int)EMBSZ) {
    // recursive insertion order
    int i = 0;
    p.emb = (const float*)d_in[i++]; p.pos = (const float*)d_in[i++];
    p.ln_g = (const float*)d_in[i++]; p.ln_b = (const float*)d_in[i++];
    for (int l = 0; l < LLAY; ++l) {
      p.Wq[l] = (const float*)d_in[i++]; p.bq[l] = (const float*)d_in[i++];
      p.Wk[l] = (const float*)d_in[i++]; p.bk[l] = (const float*)d_in[i++];
      p.Wv[l] = (const float*)d_in[i++]; p.bv[l] = (const float*)d_in[i++];
      p.Wo[l] = (const float*)d_in[i++]; p.bo[l] = (const float*)d_in[i++];
      p.ln1g[l] = (const float*)d_in[i++]; p.ln1b[l] = (const float*)d_in[i++];
      p.W1[l] = (const float*)d_in[i++]; p.b1[l] = (const float*)d_in[i++];
      p.W2[l] = (const float*)d_in[i++]; p.b2[l] = (const float*)d_in[i++];
      p.ln2g[l] = (const float*)d_in[i++]; p.ln2b[l] = (const float*)d_in[i++];
    }
    p.Wa = (const float*)d_in[i++]; p.ba = (const float*)d_in[i++];
    p.Wsp = (const float*)d_in[i++]; p.bsp = (const float*)d_in[i++];
    p.Wc = (const float*)d_in[i++]; p.bc = (const float*)d_in[i++];
    p.ids = (const int*)d_in[i++];
    p.am = (const float*)d_in[i++]; p.gm = (const float*)d_in[i++];
    p.mh = (const float*)d_in[i++]; p.mlh = (const float*)d_in[i++];
    p.mas = (const float*)d_in[i++]; p.mlc = (const float*)d_in[i++];
  } else if (in_sizes[0] == DD * 3) {
    // top-level insertion, params sorted
    int i = 0;
    read_params_sorted(d_in, i, p);
    p.ids = (const int*)d_in[i++];
    p.am = (const float*)d_in[i++]; p.gm = (const float*)d_in[i++];
    p.mh = (const float*)d_in[i++]; p.mlh = (const float*)d_in[i++];
    p.mas = (const float*)d_in[i++]; p.mlc = (const float*)d_in[i++];
  } else {
    // fully sorted
    int i = 0;
    p.am = (const float*)d_in[i++]; p.gm = (const float*)d_in[i++];
    p.ids = (const int*)d_in[i++];
    p.mas = (const float*)d_in[i++]; p.mh = (const float*)d_in[i++];
    p.mlc = (const float*)d_in[i++]; p.mlh = (const float*)d_in[i++];
    read_params_sorted(d_in, i, p);
  }

  // workspace carve-up
  char* w = (char*)d_ws;
  auto af = [&](long n) { float* q = (float*)w; w += ((n * 4 + 255) & ~255L); return q; };
  auto ah = [&](long n) { h16* q = (h16*)w; w += ((n * 2 + 255) & ~255L); return q; };
  auto ai = [&](long n) { int* q = (int*)w; w += ((n * 4 + 255) & ~255L); return q; };

  float* x      = af((long)BSTOT * DD);
  float* y      = af((long)BSTOT * DD);
  h16*   xh     = ah((long)BSTOT * DD);
  h16*   qh     = ah((long)BSTOT * DD);
  h16*   kh     = ah((long)BSTOT * DD);
  h16*   vh     = ah((long)BSTOT * DD);
  h16*   vt     = ah((long)BSTOT * DD);
  h16*   attn_h = ah((long)BSTOT * DD);
  h16*   ffh    = ah((long)BSTOT * FFD);
  float* Sg     = af((long)BB * HH * GGL * SS);
  h16*   Pg     = ah((long)BB * HH * GGL * SS);
  h16* wq_t[2], *wk_t[2], *wv_t[2], *wo_t[2], *w1_t[2], *w2_t[2];
  for (int l = 0; l < LLAY; ++l) {
    wq_t[l] = ah((long)DD * DD); wk_t[l] = ah((long)DD * DD);
    wv_t[l] = ah((long)DD * DD); wo_t[l] = ah((long)DD * DD);
    w1_t[l] = ah((long)DD * FFD); w2_t[l] = ah((long)FFD * DD);
  }
  h16*   wc_t   = ah((long)DD * DD);
  float* pe     = af((long)BSTOT * 3);
  float* ps     = af((long)BSTOT * 2);
  int*   order  = ai((long)BB * MPAD);
  float* mask_r = af((long)BB * MPAD);
  h16*   hr_h   = ah((long)BB * MPAD * DD);
  h16*   ht_h   = ah((long)BB * MPAD * DD);
  float* pc     = af((long)BB * MPAD * MPAD);
  float* acc    = af(16);

  auto gemm = [&](const h16* A, int lda, const h16* Bt, int ldb, const float* bias,
                  float* Cf, h16* Ch, int ldc, int M, int N, int K, int act) {
    int mt = (M + 15) / 16, nb = (N + 63) / 64;
    k_gemm<<<dim3(mt * nb), dim3(32), 0, stream>>>(A, lda, Bt, ldb, bias, Cf, Ch, ldc, M, N, K, act, nb);
  };
  auto w2h = [&](const float* W, h16* Wt, int K, int N) {
    long tot = (long)K * N;
    k_w2h_t<<<dim3((unsigned)((tot + 255) / 256)), dim3(256), 0, stream>>>(W, Wt, K, N);
  };

  k_zero<<<1, 32, 0, stream>>>(acc);

  for (int l = 0; l < LLAY; ++l) {
    w2h(p.Wq[l], wq_t[l], DD, DD);  w2h(p.Wk[l], wk_t[l], DD, DD);
    w2h(p.Wv[l], wv_t[l], DD, DD);  w2h(p.Wo[l], wo_t[l], DD, DD);
    w2h(p.W1[l], w1_t[l], DD, FFD); w2h(p.W2[l], w2_t[l], FFD, DD);
  }
  w2h(p.Wc, wc_t, DD, DD);

  k_embed_ln<<<dim3(BSTOT), dim3(256), 0, stream>>>(p.ids, p.emb, p.pos, p.ln_g, p.ln_b, x, xh);

  for (int l = 0; l < LLAY; ++l) {
    gemm(xh, DD, wq_t[l], DD, p.bq[l], nullptr, qh, DD, BSTOT, DD, DD, 0);
    gemm(xh, DD, wk_t[l], DD, p.bk[l], nullptr, kh, DD, BSTOT, DD, DD, 0);
    gemm(xh, DD, wv_t[l], DD, p.bv[l], nullptr, vh, DD, BSTOT, DD, DD, 0);
    k_vtrans<<<dim3((BSTOT * DD + 255) / 256), dim3(256), 0, stream>>>(vh, vt);
    k_local_attn<<<dim3(BB * HH * NCC * (CCH / 16)), dim3(32), 0, stream>>>(qh, kh, vt, p.am, p.gm, attn_h);
    k_gscore<<<dim3(BB * HH * 4 * (SS / 64)), dim3(32), 0, stream>>>(qh, kh, p.am, Sg);
    k_gsoftmax<<<dim3(BB * HH * GGL), dim3(32), 0, stream>>>(Sg, Pg);
    k_gout<<<dim3(BB * HH * 4), dim3(32), 0, stream>>>(Pg, vt, attn_h);
    gemm(attn_h, DD, wo_t[l], DD, p.bo[l], y, nullptr, DD, BSTOT, DD, DD, 0);
    k_addln<<<dim3(BSTOT), dim3(256), 0, stream>>>(x, y, p.ln1g[l], p.ln1b[l], xh);
    gemm(xh, DD, w1_t[l], DD, p.b1[l], nullptr, ffh, FFD, BSTOT, FFD, DD, 1);
    gemm(ffh, FFD, w2_t[l], FFD, p.b2[l], y, nullptr, DD, BSTOT, DD, FFD, 0);
    k_addln<<<dim3(BSTOT), dim3(256), 0, stream>>>(x, y, p.ln2g[l], p.ln2b[l], xh);
  }

  k_heads<<<dim3((BB * SS * 5 + 7) / 8), dim3(256), 0, stream>>>(x, p.Wa, p.ba, p.Wsp, p.bsp, pe, ps);
  k_count_es<<<dim3(64), dim3(256), 0, stream>>>(p.mlh, p.mas, acc);
  k_loss_es<<<dim3(64), dim3(256), 0, stream>>>(pe, ps, p.mlh, p.mas, acc);

  k_order<<<dim3(BB), dim3(32), 0, stream>>>(p.mh, order, mask_r);
  k_gather<<<dim3(BB * MPAD), dim3(256), 0, stream>>>(x, order, hr_h);
  gemm(hr_h, DD, wc_t, DD, p.bc, nullptr, ht_h, DD, BB * MPAD, DD, DD, 2);
  k_pc<<<dim3(BB * 13 * 13), dim3(32), 0, stream>>>(ht_h, hr_h, pc);
  k_count_cond<<<dim3(64), dim3(256), 0, stream>>>(mask_r, order, p.mlc, acc);
  k_loss_cond<<<dim3(64), dim3(256), 0, stream>>>(pc, mask_r, order, p.mlc, acc);

  k_finalize<<<1, 32, 0, stream>>>(acc, (float*)d_out);
}
